// BiDecoderRNN_21440476741650
// MI455X (gfx1250) — matrix operand order repaired and checked by reading, not compiled, not run on hardware
//
#include <hip/hip_runtime.h>
#include <hip/hip_bf16.h>

typedef __attribute__((ext_vector_type(16))) __bf16 v16bf;
typedef __attribute__((ext_vector_type(8)))  float  v8f;

#define S_   400
#define B_   16
#define T_   32
#define H_   256
#define H2_  512
#define A_   100
#define AP_  112
#define G_   50
#define VF_  100
#define VK_  128
#define V_   50000
#define BV_  50050
#define GATE_ 1024
#define VTILES_ 3125            /* 50000/16 */
#define VTPAD_  3128            /* 391 blocks * 8 waves */
#define PADN_   (VTPAD_ * 16)   /* padded rows of W_v2 bf16 copy */

__device__ __forceinline__ __bf16 f2bf(float f) {
  unsigned u = __builtin_bit_cast(unsigned, f);
  unsigned r = (u + 0x7FFFu + ((u >> 16) & 1u)) >> 16;
  return __builtin_bit_cast(__bf16, (unsigned short)r);
}
__device__ __forceinline__ float sigf(float x) { return 1.0f / (1.0f + expf(-x)); }

// ---- WMMA fragment loaders (gfx1250 wave32 layouts, CDNA5 ISA) ----
// A: 16x32 (MxK) bf16. lane holds row m=lane&15; 16 contiguous K starting at k0+half*16
__device__ __forceinline__ v16bf load_a_bf16(const __bf16* base, int ld, int k0, int lane) {
  int m = lane & 15, half = (lane >> 4) & 1;
  const __bf16* p = base + (size_t)m * ld + k0 + half * 16;
  v16bf a;
#pragma unroll
  for (int i = 0; i < 8; ++i) a[i] = p[i];
#pragma unroll
  for (int i = 0; i < 8; ++i) a[i + 8] = p[i + 8];
  return a;
}
__device__ __forceinline__ v16bf load_a_f32(const float* base, int ld, int k0, int lane) {
  int m = lane & 15, half = (lane >> 4) & 1;
  const float* p = base + (size_t)m * ld + k0 + half * 16;
  v16bf a;
#pragma unroll
  for (int i = 0; i < 8; ++i) a[i] = f2bf(p[i]);
#pragma unroll
  for (int i = 0; i < 8; ++i) a[i + 8] = f2bf(p[i + 8]);
  return a;
}
// B: 32x16 (KxN) bf16, source stored row-major [N][K] (K contiguous).
// lane holds column n=lane&15, 16 contiguous K starting at k0+half*16.
__device__ __forceinline__ v16bf load_b_bf16(const __bf16* base, int ld, int k0, int lane) {
  int n = lane & 15, half = (lane >> 4) & 1;
  const __bf16* p = base + (size_t)n * ld + k0 + half * 16;
  v16bf b;
#pragma unroll
  for (int i = 0; i < 16; ++i) b[i] = p[i];
  return b;
}
__device__ __forceinline__ v8f wmma_bf16(v16bf a, v16bf b, v8f c) {
  return __builtin_amdgcn_wmma_f32_16x16x32_bf16(false, a, false, b, (short)0, c, false, false);
}

// ---- prep: bf16 weight/activation copies + zero coverage ----
__global__ void k_prep(const float* W_v2, const float* W_attn, const float* outenc,
                       const float* w_hh0, const float* w_ih1, const float* w_hh1,
                       __bf16* wv2b, __bf16* waeb, __bf16* encb,
                       __bf16* whh0b, __bf16* wih1b, __bf16* whh1b, float* cov) {
  size_t stride = (size_t)gridDim.x * blockDim.x;
  size_t tid = (size_t)blockIdx.x * blockDim.x + threadIdx.x;
  for (size_t i = tid; i < (size_t)PADN_ * VK_; i += stride) {
    int v = (int)(i >> 7), k = (int)(i & 127);
    wv2b[i] = (v < V_ && k < VF_) ? f2bf(W_v2[(size_t)v * VF_ + k]) : f2bf(0.f);
  }
  for (size_t i = tid; i < (size_t)AP_ * H2_; i += stride) {
    int a = (int)(i >> 9), k = (int)(i & 511);
    waeb[i] = (a < A_) ? f2bf(W_attn[a * 769 + k]) : f2bf(0.f);
  }
  for (size_t i = tid; i < (size_t)S_ * B_ * H2_; i += stride) encb[i] = f2bf(outenc[i]);
  for (size_t i = tid; i < (size_t)GATE_ * H_; i += stride) {
    whh0b[i] = f2bf(w_hh0[i]); wih1b[i] = f2bf(w_ih1[i]); whh1b[i] = f2bf(w_hh1[i]);
  }
  for (size_t i = tid; i < (size_t)S_ * B_; i += stride) cov[i] = 0.f;
}

// ---- encoder-state reduce: new = old_enc @ W_reduce.T + b ----
__global__ void k_reduce(const float* hid_enc, const float* cell_enc,
                         const float* W_reduce, const float* b_reduce,
                         float* nenc, float* ncell) {
  int e = blockIdx.x * blockDim.x + threadIdx.x;   // [2][16][256]
  if (e >= 2 * B_ * H_) return;
  int h = e & (H_ - 1);
  int b = (e >> 8) & (B_ - 1);
  int l = e >> 12;
  const float* wr = W_reduce + (size_t)h * H2_;
  const float* he0 = hid_enc + ((size_t)l * B_ + b) * H_;
  const float* he1 = hid_enc + ((size_t)(l + 2) * B_ + b) * H_;
  const float* ce0 = cell_enc + ((size_t)l * B_ + b) * H_;
  const float* ce1 = cell_enc + ((size_t)(l + 2) * B_ + b) * H_;
  float ah = b_reduce[h], ac = b_reduce[h];
  for (int k = 0; k < H_; ++k) { ah += wr[k] * he0[k]; ac += wr[k] * ce0[k]; }
  for (int k = 0; k < H_; ++k) { ah += wr[H_ + k] * he1[k]; ac += wr[H_ + k] * ce1[k]; }
  nenc[e] = ah; ncell[e] = ac;
}

// ---- time-invariant attention projection: eproj = output_enc @ W_attn[:, :512].T ----
// M=6400 (400 tiles), N=112 (7 tiles, padded), K=512. One tile per wave, 2800 waves.
__global__ void __launch_bounds__(256) k_encproj(const __bf16* encb, const __bf16* waeb, float* eproj) {
  int w = blockIdx.x * 8 + (threadIdx.x >> 5);
  int lane = threadIdx.x & 31;
  int mt = w / 7, nt = w - mt * 7;
  const __bf16* Ab = encb + (size_t)mt * 16 * H2_;
  const __bf16* Bb = waeb + (size_t)nt * 16 * H2_;
  v8f c = {};
  for (int k0 = 0; k0 < H2_; k0 += 32)
    c = wmma_bf16(load_a_bf16(Ab, H2_, k0, lane), load_b_bf16(Bb, H2_, k0, lane), c);
  int n = nt * 16 + (lane & 15);
  int mb = (lane >> 4) * 8;
#pragma unroll
  for (int r = 0; r < 8; ++r)
    eproj[(size_t)(mt * 16 + mb + r) * AP_ + n] = c[r];
}

// ---- whole 2-layer LSTM recurrence in one persistent workgroup (8 waves) ----
__global__ void __launch_bounds__(256) k_lstm(
    const float* input_dec, const float* nenc, const float* ncell,
    const __bf16* whh0b, const __bf16* wih1b, const __bf16* whh1b,
    const float* w_ih0, const float* b_ih0, const float* b_hh0,
    const float* b_ih1, const float* b_hh1,
    float* hstate, float* cstate, float* gates, float* odbuf,
    float* out_hid, float* out_cell) {
  int tid = threadIdx.x;
  int lane = tid & 31, wv = tid >> 5;
  for (int i = tid; i < 2 * B_ * H_; i += 256) { hstate[i] = nenc[i]; cstate[i] = ncell[i]; }
  __syncthreads();
  for (int t = 0; t < T_; ++t) {
    // layer 0: gates = h0 @ w_hh0.T  (M=16,N=1024,K=256)
    for (int tile = wv; tile < 64; tile += 8) {
      v8f c = {};
      const __bf16* Bb = whh0b + (size_t)tile * 16 * H_;
      for (int k0 = 0; k0 < H_; k0 += 32)
        c = wmma_bf16(load_a_f32(hstate, H_, k0, lane), load_b_bf16(Bb, H_, k0, lane), c);
      int n = tile * 16 + (lane & 15);
      int mb = (lane >> 4) * 8;
#pragma unroll
      for (int r = 0; r < 8; ++r) gates[(mb + r) * GATE_ + n] = c[r];
    }
    __syncthreads();
    for (int i = tid; i < B_ * H_; i += 256) {
      int b = i >> 8, hh = i & 255;
      float xb = input_dec[t * B_ + b];
      const float* g = gates + b * GATE_;
      float gi = g[hh]          + xb * w_ih0[hh]          + b_ih0[hh]          + b_hh0[hh];
      float gf = g[H_ + hh]     + xb * w_ih0[H_ + hh]     + b_ih0[H_ + hh]     + b_hh0[H_ + hh];
      float gg = g[2 * H_ + hh] + xb * w_ih0[2 * H_ + hh] + b_ih0[2 * H_ + hh] + b_hh0[2 * H_ + hh];
      float go = g[3 * H_ + hh] + xb * w_ih0[3 * H_ + hh] + b_ih0[3 * H_ + hh] + b_hh0[3 * H_ + hh];
      float cc = sigf(gf) * cstate[i] + sigf(gi) * tanhf(gg);
      cstate[i] = cc;
      hstate[i] = sigf(go) * tanhf(cc);
    }
    __syncthreads();
    // layer 1: gates = y0 @ w_ih1.T + h1 @ w_hh1.T
    for (int tile = wv; tile < 64; tile += 8) {
      v8f c = {};
      const __bf16* B1 = wih1b + (size_t)tile * 16 * H_;
      const __bf16* B2 = whh1b + (size_t)tile * 16 * H_;
      for (int k0 = 0; k0 < H_; k0 += 32)
        c = wmma_bf16(load_a_f32(hstate, H_, k0, lane), load_b_bf16(B1, H_, k0, lane), c);
      for (int k0 = 0; k0 < H_; k0 += 32)
        c = wmma_bf16(load_a_f32(hstate + B_ * H_, H_, k0, lane), load_b_bf16(B2, H_, k0, lane), c);
      int n = tile * 16 + (lane & 15);
      int mb = (lane >> 4) * 8;
#pragma unroll
      for (int r = 0; r < 8; ++r) gates[(mb + r) * GATE_ + n] = c[r];
    }
    __syncthreads();
    for (int i = tid; i < B_ * H_; i += 256) {
      int b = i >> 8, hh = i & 255;
      const float* g = gates + b * GATE_;
      float gi = g[hh]          + b_ih1[hh]          + b_hh1[hh];
      float gf = g[H_ + hh]     + b_ih1[H_ + hh]     + b_hh1[H_ + hh];
      float gg = g[2 * H_ + hh] + b_ih1[2 * H_ + hh] + b_hh1[2 * H_ + hh];
      float go = g[3 * H_ + hh] + b_ih1[3 * H_ + hh] + b_hh1[3 * H_ + hh];
      int j = B_ * H_ + i;
      float cc = sigf(gf) * cstate[j] + sigf(gi) * tanhf(gg);
      cstate[j] = cc;
      float hv = sigf(go) * tanhf(cc);
      hstate[j] = hv;
      odbuf[(size_t)t * B_ * H_ + i] = hv;     // output_dec
    }
    __syncthreads();
  }
  for (int i = tid; i < 2 * B_ * H_; i += 256) { out_hid[i] = hstate[i]; out_cell[i] = cstate[i]; }
}

// ---- per-step attention + softmax + coverage + context + pgen + V1 hidden ----
__global__ void __launch_bounds__(256) k_att(
    int t, const float* outenc, const float* W_attn, const float* b_attn,
    const float* w_attn_out, const float* att_mask, const float* input_dec,
    const float* embed, const float* W_pgen, const float* b_pgen,
    const float* W_v1, const float* b_v1, const float* odbuf, const float* eproj,
    float* cov, float* attnbuf, float* pgenbuf, __bf16* hidb, float* out_cl) {
  int b = blockIdx.x, tid = threadIdx.x;
  __shared__ float od[H_];
  __shared__ float odp[AP_];
  __shared__ float ee[S_];
  __shared__ float ctx[H2_];
  __shared__ float red[256];
  __shared__ float sc[2];
  for (int i = tid; i < H_; i += 256) od[i] = odbuf[((size_t)t * B_ + b) * H_ + i];
  __syncthreads();
  // decoder-state projection + bias: odp[a] = b_attn[a] + W_attn[a,512:768] . od
  for (int a = tid; a < AP_; a += 256) {
    float acc = 0.f;
    if (a < A_) {
      acc = b_attn[a];
      const float* w = W_attn + a * 769 + H2_;
      for (int k = 0; k < H_; ++k) acc += w[k] * od[k];
    }
    odp[a] = acc;
  }
  __syncthreads();
  // e[s] = w_attn_out . tanh(eproj + odp + w_cov*cov) + att_mask
  for (int s = tid; s < S_; s += 256) {
    float cv = cov[s * B_ + b];
    const float* ep = eproj + (size_t)(s * B_ + b) * AP_;
    float acc = 0.f;
    for (int a = 0; a < A_; ++a) {
      float z = ep[a] + odp[a] + W_attn[a * 769 + 768] * cv;
      acc += w_attn_out[a] * tanhf(z);
    }
    ee[s] = acc + att_mask[s * B_ + b];
  }
  __syncthreads();
  // softmax over S
  float m = -1e30f;
  for (int s = tid; s < S_; s += 256) m = fmaxf(m, ee[s]);
  red[tid] = m; __syncthreads();
  for (int k = 128; k; k >>= 1) { if (tid < k) red[tid] = fmaxf(red[tid], red[tid + k]); __syncthreads(); }
  if (tid == 0) sc[0] = red[0];
  __syncthreads();
  float sum = 0.f;
  for (int s = tid; s < S_; s += 256) { float v = expf(ee[s] - sc[0]); ee[s] = v; sum += v; }
  red[tid] = sum; __syncthreads();
  for (int k = 128; k; k >>= 1) { if (tid < k) red[tid] += red[tid + k]; __syncthreads(); }
  if (tid == 0) sc[1] = red[0];
  __syncthreads();
  float inv = 1.0f / sc[1];
  float cl = 0.f;
  for (int s = tid; s < S_; s += 256) {
    float at = ee[s] * inv;
    float co = cov[s * B_ + b];
    cl += fminf(at, co);           // coverage loss uses pre-update coverage
    cov[s * B_ + b] = co + at;
    attnbuf[s * B_ + b] = at;
    ee[s] = at;
  }
  red[tid] = cl; __syncthreads();
  for (int k = 128; k; k >>= 1) { if (tid < k) red[tid] += red[tid + k]; __syncthreads(); }
  if (tid == 0) out_cl[b] = red[0];
  __syncthreads();
  // context = sum_s attn * output_enc
  for (int ch = tid; ch < H2_; ch += 256) {
    float acc = 0.f;
    for (int s = 0; s < S_; ++s) acc += ee[s] * outenc[((size_t)s * B_ + b) * H2_ + ch];
    ctx[ch] = acc;
  }
  __syncthreads();
  // pgen = sigmoid(W_pgen . [context, od, emb] + b)
  int tok = (int)input_dec[t * B_ + b];
  const float* er = embed + (size_t)tok * G_;
  float p = 0.f;
  for (int k = tid; k < 818; k += 256) {
    float xv = (k < H2_) ? ctx[k] : ((k < 768) ? od[k - H2_] : er[k - 768]);
    p += W_pgen[k] * xv;
  }
  red[tid] = p; __syncthreads();
  for (int k = 128; k; k >>= 1) { if (tid < k) red[tid] += red[tid + k]; __syncthreads(); }
  if (tid == 0) pgenbuf[b] = 1.0f / (1.0f + expf(-(red[0] + b_pgen[0])));
  __syncthreads();
  // hid = W_v1 @ [od, context] + b_v1  -> bf16, K padded to 128
  for (int j = tid; j < VK_; j += 256) {
    float acc = 0.f;
    if (j < VF_) {
      acc = b_v1[j];
      const float* w = W_v1 + j * 768;
      for (int k = 0; k < H_; ++k) acc += w[k] * od[k];
      for (int k = 0; k < H2_; ++k) acc += w[H_ + k] * ctx[k];
    }
    hidb[b * VK_ + j] = f2bf(acc);
  }
}

// ---- vocab GEMM: logits[16,50000] = hid @ W_v2.T + b_v2 ----
// Per block: 8 waves consume an 8-tile (32 KB) B-panel of W_v2, staged HBM/L2 -> LDS
// with GLOBAL_LOAD_ASYNC_TO_LDS_B128 (ASYNCcnt path), then bf16 WMMA from LDS.
__global__ void __launch_bounds__(256) k_vgemm(const __bf16* hidb, const __bf16* wv2b,
                                               const float* b_v2, float* logits) {
  __shared__ __align__(16) __bf16 Bs[8 * 16 * VK_];   // 32 KB
  int tile0 = blockIdx.x * 8;
  const __bf16* gB = wv2b + (size_t)tile0 * 16 * VK_;
  unsigned lds_base = (unsigned)(size_t)&Bs[0];
#pragma unroll
  for (int it = 0; it < 8; ++it) {
    int idx = it * 256 + threadIdx.x;                       // 2048 x 16B = 32 KB
    unsigned lds_off = lds_base + idx * 16;
    unsigned long long ga = (unsigned long long)(size_t)gB + (unsigned long long)idx * 16;
    asm volatile("global_load_async_to_lds_b128 %0, %1, off"
                 :: "v"(lds_off), "v"(ga) : "memory");
  }
  asm volatile("s_wait_asynccnt 0x0" ::: "memory");
  __syncthreads();
  int wv = threadIdx.x >> 5, lane = threadIdx.x & 31;
  int w = tile0 + wv;
  int n = lane & 15, half = (lane >> 4) & 1;
  v8f c = {};
#pragma unroll
  for (int k0 = 0; k0 < VK_; k0 += 32) {
    v16bf a = load_a_bf16(hidb, VK_, k0, lane);
    v16bf bfr;
    const __bf16* p = &Bs[(wv * 16 + n) * VK_ + k0 + half * 16];
#pragma unroll
    for (int i = 0; i < 16; ++i) bfr[i] = p[i];
    c = wmma_bf16(a, bfr, c);
  }
  if (w < VTILES_) {                 // wave-uniform guard (EXEC stayed all-ones for WMMA)
    int nn = w * 16 + n;
    float bias = b_v2[nn];
    int mb = half * 8;
#pragma unroll
    for (int r = 0; r < 8; ++r) logits[(size_t)(mb + r) * V_ + nn] = c[r] + bias;
  }
}

// ---- softmax stats over V per batch row ----
__global__ void __launch_bounds__(256) k_smax(const float* logits, float* smax, float* ssum) {
  __shared__ float red[256];
  int b = blockIdx.x, tid = threadIdx.x;
  const float* row = logits + (size_t)b * V_;
  float m = -1e30f;
  for (int i = tid; i < V_; i += 256) m = fmaxf(m, row[i]);
  red[tid] = m; __syncthreads();
  for (int k = 128; k; k >>= 1) { if (tid < k) red[tid] = fmaxf(red[tid], red[tid + k]); __syncthreads(); }
  m = red[0]; __syncthreads();
  float s = 0.f;
  for (int i = tid; i < V_; i += 256) s += expf(row[i] - m);
  red[tid] = s; __syncthreads();
  for (int k = 128; k; k >>= 1) { if (tid < k) red[tid] += red[tid + k]; __syncthreads(); }
  if (tid == 0) { smax[b] = m; ssum[b] = red[0]; }
}

// ---- write pvocab row (softmax * pgen, zero-pad to BV) ----
__global__ void k_write(const float* logits, const float* smax, const float* ssum,
                        const float* pgen, float* out_pv) {
  size_t i = (size_t)blockIdx.x * 256 + threadIdx.x;
  if (i >= (size_t)B_ * BV_) return;
  int b = (int)(i / BV_);
  int v = (int)(i - (size_t)b * BV_);
  float val = 0.f;
  if (v < V_) val = expf(logits[(size_t)b * V_ + v] - smax[b]) / ssum[b] * pgen[b];
  out_pv[i] = val;
}

// ---- scatter-add copy distribution ----
__global__ void k_scatter(const int* real_index, const float* attn, const float* pgen, float* out_pv) {
  int i = blockIdx.x * 256 + threadIdx.x;
  if (i >= S_ * B_) return;
  int b = i & 15;
  atomicAdd(out_pv + (size_t)b * BV_ + real_index[i], (1.0f - pgen[b]) * attn[i]);
}

__global__ void k_covout(const float* cov, float* out_cov) {
  int i = blockIdx.x * 256 + threadIdx.x;
  if (i < S_ * B_) out_cov[i] = cov[i];
}

extern "C" void kernel_launch(void* const* d_in, const int* in_sizes, int n_in,
                              void* d_out, int out_size, void* d_ws, size_t ws_size,
                              hipStream_t stream) {
  (void)in_sizes; (void)n_in; (void)out_size; (void)ws_size;
  const float* outenc     = (const float*)d_in[0];
  const int*   real_index = (const int*)d_in[1];
  const float* input_dec  = (const float*)d_in[3];
  const float* hidden_enc = (const float*)d_in[4];
  const float* cell_enc   = (const float*)d_in[5];
  const float* att_mask   = (const float*)d_in[6];
  const float* embed      = (const float*)d_in[7];
  const float* W_reduce   = (const float*)d_in[8];
  const float* b_reduce   = (const float*)d_in[9];
  const float* w_ih0      = (const float*)d_in[10];
  const float* w_hh0      = (const float*)d_in[11];
  const float* b_ih0      = (const float*)d_in[12];
  const float* b_hh0      = (const float*)d_in[13];
  const float* w_ih1      = (const float*)d_in[14];
  const float* w_hh1      = (const float*)d_in[15];
  const float* b_ih1      = (const float*)d_in[16];
  const float* b_hh1      = (const float*)d_in[17];
  const float* W_attn     = (const float*)d_in[18];
  const float* b_attn     = (const float*)d_in[19];
  const float* w_attn_out = (const float*)d_in[20];
  const float* W_v1       = (const float*)d_in[21];
  const float* b_v1       = (const float*)d_in[22];
  const float* W_v2       = (const float*)d_in[23];
  const float* b_v2       = (const float*)d_in[24];
  const float* W_pgen     = (const float*)d_in[25];
  const float* b_pgen     = (const float*)d_in[26];

  char* ws = (char*)d_ws;
  size_t off = 0;
  auto alloc = [&](size_t bytes) { size_t o = off; off = (off + bytes + 255) & ~(size_t)255; return o; };
  __bf16* wv2b   = (__bf16*)(ws + alloc((size_t)PADN_ * VK_ * 2));
  __bf16* waeb   = (__bf16*)(ws + alloc((size_t)AP_ * H2_ * 2));
  __bf16* encb   = (__bf16*)(ws + alloc((size_t)S_ * B_ * H2_ * 2));
  float*  eproj  = (float*) (ws + alloc((size_t)S_ * B_ * AP_ * 4));
  __bf16* whh0b  = (__bf16*)(ws + alloc((size_t)GATE_ * H_ * 2));
  __bf16* wih1b  = (__bf16*)(ws + alloc((size_t)GATE_ * H_ * 2));
  __bf16* whh1b  = (__bf16*)(ws + alloc((size_t)GATE_ * H_ * 2));
  float*  nenc   = (float*) (ws + alloc((size_t)2 * B_ * H_ * 4));
  float*  ncell  = (float*) (ws + alloc((size_t)2 * B_ * H_ * 4));
  float*  hstate = (float*) (ws + alloc((size_t)2 * B_ * H_ * 4));
  float*  cstate = (float*) (ws + alloc((size_t)2 * B_ * H_ * 4));
  float*  gates  = (float*) (ws + alloc((size_t)B_ * GATE_ * 4));
  float*  odbuf  = (float*) (ws + alloc((size_t)T_ * B_ * H_ * 4));
  float*  cov    = (float*) (ws + alloc((size_t)S_ * B_ * 4));
  float*  attnbf = (float*) (ws + alloc((size_t)S_ * B_ * 4));
  float*  pgenbf = (float*) (ws + alloc((size_t)B_ * 4));
  __bf16* hidb   = (__bf16*)(ws + alloc((size_t)B_ * VK_ * 2));
  float*  logits = (float*) (ws + alloc((size_t)B_ * V_ * 4));
  float*  smaxb  = (float*) (ws + alloc((size_t)B_ * 4));
  float*  ssumb  = (float*) (ws + alloc((size_t)B_ * 4));

  float* out = (float*)d_out;
  const size_t OFF_CL   = (size_t)T_ * B_ * BV_;
  const size_t OFF_COV  = OFF_CL + (size_t)T_ * B_;
  const size_t OFF_HID  = OFF_COV + (size_t)S_ * B_;
  const size_t OFF_CELL = OFF_HID + (size_t)2 * B_ * H_;

  k_prep<<<1024, 256, 0, stream>>>(W_v2, W_attn, outenc, w_hh0, w_ih1, w_hh1,
                                   wv2b, waeb, encb, whh0b, wih1b, whh1b, cov);
  k_reduce<<<(2 * B_ * H_ + 255) / 256, 256, 0, stream>>>(hidden_enc, cell_enc,
                                                          W_reduce, b_reduce, nenc, ncell);
  k_encproj<<<350, 256, 0, stream>>>(encb, waeb, eproj);
  k_lstm<<<1, 256, 0, stream>>>(input_dec, nenc, ncell, whh0b, wih1b, whh1b,
                                w_ih0, b_ih0, b_hh0, b_ih1, b_hh1,
                                hstate, cstate, gates, odbuf,
                                out + OFF_HID, out + OFF_CELL);
  for (int t = 0; t < T_; ++t) {
    k_att<<<B_, 256, 0, stream>>>(t, outenc, W_attn, b_attn, w_attn_out, att_mask,
                                  input_dec, embed, W_pgen, b_pgen, W_v1, b_v1,
                                  odbuf, eproj, cov, attnbf, pgenbf, hidb,
                                  out + OFF_CL + (size_t)t * B_);
    k_vgemm<<<VTPAD_ / 8, 256, 0, stream>>>(hidb, wv2b, b_v2, logits);
    k_smax<<<B_, 256, 0, stream>>>(logits, smaxb, ssumb);
    float* out_pv = out + (size_t)t * B_ * BV_;
    k_write<<<(int)(((size_t)B_ * BV_ + 255) / 256), 256, 0, stream>>>(logits, smaxb, ssumb,
                                                                       pgenbf, out_pv);
    k_scatter<<<(S_ * B_ + 255) / 256, 256, 0, stream>>>(real_index, attnbf, pgenbf, out_pv);
  }
  k_covout<<<(S_ * B_ + 255) / 256, 256, 0, stream>>>(cov, out + OFF_COV);
}
